// CIKM20_TK_Sparse_34084860461081
// MI455X (gfx1250) — matrix operand order repaired and checked by reading, not compiled, not run on hardware
//
#include <hip/hip_runtime.h>
#include <hip/hip_bf16.h>

// ---------------------------------------------------------------------------
// CIKM20 TK (Transformer-Kernel) ranking model forward for MI455X (gfx1250).
// bf16 WMMA (v_wmma_f32_16x16x32_bf16) for all GEMMs + flash attention,
// f32 statistics for LayerNorm / softmax / kernel pooling.
// ---------------------------------------------------------------------------

typedef unsigned short bf16_t;
typedef __attribute__((ext_vector_type(16))) __bf16 v16bf;
typedef __attribute__((ext_vector_type(8)))  __bf16 v8bf;
typedef __attribute__((ext_vector_type(8)))  float  v8f;

#define B_SZ   256
#define Q_LEN  32
#define D_LEN  512
#define E_DIM  256
#define FF_DIM 1024
#define NHEAD  8
#define DHEAD  32
#define NKER   11
#define TQ     (B_SZ * Q_LEN)   // 8192 query tokens
#define TD     (B_SZ * D_LEN)   // 131072 document tokens

__device__ __forceinline__ bf16_t f2bf(float x) {
    unsigned u = __float_as_uint(x);
    unsigned r = u + 0x7FFFu + ((u >> 16) & 1u);
    return (bf16_t)(r >> 16);
}

__device__ __forceinline__ float wred_add16(float x) {
    x += __shfl_xor(x, 1, 32);
    x += __shfl_xor(x, 2, 32);
    x += __shfl_xor(x, 4, 32);
    x += __shfl_xor(x, 8, 32);
    return x;
}
__device__ __forceinline__ float wred_max16(float x) {
    x = fmaxf(x, __shfl_xor(x, 1, 32));
    x = fmaxf(x, __shfl_xor(x, 2, 32));
    x = fmaxf(x, __shfl_xor(x, 4, 32));
    x = fmaxf(x, __shfl_xor(x, 8, 32));
    return x;
}
__device__ __forceinline__ float wred_add32(float x) {
    x = wred_add16(x);
    x += __shfl_xor(x, 16, 32);
    return x;
}

// load a 16-element bf16 fragment from LDS as two 16-byte chunks
__device__ __forceinline__ v16bf ldfrag(const bf16_t* p0, const bf16_t* p1) {
    v8bf a = *(const v8bf*)p0;
    v8bf b = *(const v8bf*)p1;
    v16bf r;
#pragma unroll
    for (int i = 0; i < 8; ++i) { r[i] = a[i]; r[i + 8] = b[i]; }
    return r;
}

// ---------------------------------------------------------------------------
// f32 -> bf16 conversion (weights)
// ---------------------------------------------------------------------------
__global__ void f2bf_k(const float* __restrict__ src, bf16_t* __restrict__ dst,
                       long long n) {
    long long i = (long long)blockIdx.x * 256 + threadIdx.x;
    if (i < n) dst[i] = f2bf(src[i]);
}

// ---------------------------------------------------------------------------
// sinusoidal positional encoding: pos[t, 0..127]=sin(t*inv_j), [128..255]=cos
// ---------------------------------------------------------------------------
__global__ void posenc_k(float* __restrict__ pos) {
    int i = blockIdx.x * 256 + threadIdx.x;      // 512 * 128 elements
    if (i >= 512 * 128) return;
    int t = i >> 7, j = i & 127;
    float inv = expf((float)j * (-9.210340371976184f / 127.0f));
    float st = (float)t * inv;
    pos[t * E_DIM + j] = sinf(st);
    pos[t * E_DIM + 128 + j] = cosf(st);
}

// ---------------------------------------------------------------------------
// h = emb*mask + pos ; also bf16 copy
// ---------------------------------------------------------------------------
__global__ void prep_k(const float* __restrict__ emb, const float* __restrict__ mask,
                       const float* __restrict__ pos, float* __restrict__ h,
                       bf16_t* __restrict__ hbf, int S, long long total) {
    long long i = (long long)blockIdx.x * 256 + threadIdx.x;
    if (i >= total) return;
    int e = (int)(i & (E_DIM - 1));
    long long tok = i >> 8;
    int s = (int)(tok % S);
    float x = emb[i] * mask[tok];
    float v = x + pos[s * E_DIM + e];
    h[i] = v;
    hbf[i] = f2bf(v);
}

// ---------------------------------------------------------------------------
// Generic bf16 GEMM:  C[M,N] = act(A[M,K] @ W[N,K]^T + bias)
// block = 256 threads (8 waves); tile M=128, N=64, K-step 32.
// batched via z strides (elements). act: 0=none 1=relu 2=tanh.
// All fragments are fetched before the 4 back-to-back WMMAs so DS loads and
// XDL WMMA execution can overlap; next K-tile is prefetched to L2.
// ---------------------------------------------------------------------------
__global__ __launch_bounds__(256)
void gemm_bt(const bf16_t* __restrict__ A, long long sAb, int lda,
             const bf16_t* __restrict__ Bm, long long sBb, int ldb,
             const float* __restrict__ bias,
             float* __restrict__ Cf, bf16_t* __restrict__ Cb, long long sCb,
             int ldc, int M, int N, int K, int act) {
    __shared__ __align__(64) bf16_t As[128][32];
    __shared__ __align__(64) bf16_t Bs[64][32];

    const int tid = threadIdx.x;
    const int wave = tid >> 5, lane = tid & 31;
    const int g = lane >> 4, ln = lane & 15;
    const int m0 = blockIdx.y * 128;
    const int n0 = blockIdx.x * 64;
    const long long zb = blockIdx.z;
    const bf16_t* Ab = A + zb * sAb;
    const bf16_t* Bb = Bm + zb * sBb;

    v8f acc[4];
#pragma unroll
    for (int t = 0; t < 4; ++t)
#pragma unroll
        for (int i = 0; i < 8; ++i) acc[t][i] = 0.0f;

    for (int k0 = 0; k0 < K; k0 += 32) {
        {   // stage A tile (bounds-checked on M)
            int r = tid >> 1, half = tid & 1;
            uint4 z0 = make_uint4(0, 0, 0, 0), z1 = make_uint4(0, 0, 0, 0);
            if (m0 + r < M) {
                const uint4* src =
                    (const uint4*)(Ab + (long long)(m0 + r) * lda + k0 + half * 16);
                z0 = src[0];
                z1 = src[1];
            }
            uint4* dst = (uint4*)(&As[r][half * 16]);
            dst[0] = z0;
            dst[1] = z1;
        }
        {   // stage B tile (bounds-checked on N)
            int r = tid >> 2, qq = tid & 3;
            uint4 z = make_uint4(0, 0, 0, 0);
            if (n0 + r < N)
                z = *(const uint4*)(Bb + (long long)(n0 + r) * ldb + k0 + qq * 8);
            *(uint4*)(&Bs[r][qq * 8]) = z;
        }
        // prefetch next K-tile toward L2 while this tile computes
        if (k0 + 32 < K) {
            int ra = tid >> 1;
            if (m0 + ra < M)
                __builtin_prefetch(Ab + (long long)(m0 + ra) * lda + k0 + 32, 0, 1);
            int rb = tid >> 2;
            if (n0 + rb < N)
                __builtin_prefetch(Bb + (long long)(n0 + rb) * ldb + k0 + 32, 0, 1);
        }
        __syncthreads();

        // fetch ALL fragments first (one DS clause), then 4 dependency-free WMMAs
        const bf16_t* ar = &As[wave * 16 + ln][0];
        v16bf av = ldfrag(ar + g * 16, ar + g * 16 + 8);
        v16bf bv[4];
#pragma unroll
        for (int t = 0; t < 4; ++t) {
            const bf16_t* br = &Bs[t * 16 + ln][g * 16];
            bv[t] = ldfrag(br, br + 8);
        }
#pragma unroll
        for (int t = 0; t < 4; ++t)
            acc[t] = __builtin_amdgcn_wmma_f32_16x16x32_bf16(
                false, av, false, bv[t], (short)0, acc[t], false, false);
        __syncthreads();
    }

    float* Cfb = Cf ? Cf + zb * sCb : (float*)0;
    bf16_t* Cbb = Cb ? Cb + zb * sCb : (bf16_t*)0;
#pragma unroll
    for (int t = 0; t < 4; ++t) {
        int col = n0 + t * 16 + ln;
        if (col >= N) continue;
        float bvl = bias ? bias[col] : 0.0f;
#pragma unroll
        for (int r = 0; r < 8; ++r) {
            int row = m0 + wave * 16 + 8 * g + r;
            if (row >= M) continue;
            float v = acc[t][r] + bvl;
            if (act == 1) v = fmaxf(v, 0.0f);
            else if (act == 2) v = tanhf(v);
            long long idx = (long long)row * ldc + col;
            if (Cfb) Cfb[idx] = v;
            if (Cbb) Cbb[idx] = f2bf(v);
        }
    }
}

// ---------------------------------------------------------------------------
// Flash attention: 4 waves per workgroup share one LDS-staged K/V tile.
// Each wave owns one 16-query-row tile (qt = blockIdx.x*4 + wave).
// qkv bf16 [T,768] (q|k|v, 8 heads x 32). ctx bf16 [T,256].
// ---------------------------------------------------------------------------
__global__ __launch_bounds__(128)
void attn_flash(const bf16_t* __restrict__ qkv, const float* __restrict__ mask,
                bf16_t* __restrict__ ctx, int S) {
    __shared__ __align__(64) bf16_t Qs[4][16][32];
    __shared__ __align__(64) bf16_t Ks[32][32];
    __shared__ __align__(64) bf16_t Vt[32][32];   // transposed: [dh][key]
    __shared__ __align__(64) bf16_t Ps[4][16][32];

    const int tid = threadIdx.x;
    const int w = tid >> 5, lane = tid & 31;
    const int g = lane >> 4, ln = lane & 15;
    const int qt = blockIdx.x * 4 + w;
    const int hh = blockIdx.y, b = blockIdx.z;
    const bool active = (qt * 16) < S;
    const long long tok0 = (long long)b * S;

    if (active) {   // stage this wave's Q tile (16 rows of 32 dh)
        int r = lane >> 1, half = lane & 1;
        const uint4* src =
            (const uint4*)(qkv + (tok0 + qt * 16 + r) * 768 + hh * 32 + half * 16);
        uint4* dst = (uint4*)(&Qs[w][r][half * 16]);
        dst[0] = src[0];
        dst[1] = src[1];
    }
    __syncthreads();

    const bf16_t* qr = &Qs[w][ln][0];
    v16bf qf = ldfrag(qr + g * 16, qr + g * 16 + 8);

    float mi[8], li[8];
    v8f acc0, acc1;
#pragma unroll
    for (int r = 0; r < 8; ++r) {
        mi[r] = -1e30f; li[r] = 0.0f; acc0[r] = 0.0f; acc1[r] = 0.0f;
    }
    const float scale = 0.17677669529663687f;   // 1/sqrt(32)

    for (int kt = 0; kt < S; kt += 32) {
        __syncthreads();   // previous iteration's Ks/Vt/Ps reads complete
        {   // stage K tile (natural [key][dh]) -- 128 threads, one uint4 each
            int r = tid >> 2, qq = tid & 3;
            *(uint4*)(&Ks[r][qq * 8]) =
                *(const uint4*)(qkv + (tok0 + kt + r) * 768 + 256 + hh * 32 + qq * 8);
        }
        {   // stage V transposed ([dh][key]) -- 8 elements per thread
            int r = tid & 31, qq = tid >> 5;
            const bf16_t* src = qkv + (tok0 + kt + r) * 768 + 512 + hh * 32 + qq * 8;
#pragma unroll
            for (int d2 = 0; d2 < 8; ++d2) Vt[qq * 8 + d2][r] = src[d2];
        }
        __syncthreads();

        if (active) {
            // both K fragments first, then two back-to-back score WMMAs
            const bf16_t* k0p = &Ks[ln][g * 16];
            const bf16_t* k1p = &Ks[16 + ln][g * 16];
            v16bf kf0 = ldfrag(k0p, k0p + 8);
            v16bf kf1 = ldfrag(k1p, k1p + 8);
            v8f s0, s1;
#pragma unroll
            for (int i = 0; i < 8; ++i) { s0[i] = 0.0f; s1[i] = 0.0f; }
            s0 = __builtin_amdgcn_wmma_f32_16x16x32_bf16(
                false, qf, false, kf0, (short)0, s0, false, false);
            s1 = __builtin_amdgcn_wmma_f32_16x16x32_bf16(
                false, qf, false, kf1, (short)0, s1, false, false);

            float km0 = mask[(long long)b * S + kt + ln];
            float km1 = mask[(long long)b * S + kt + 16 + ln];

#pragma unroll
            for (int r = 0; r < 8; ++r) {
                float x0 = (km0 > 0.0f) ? s0[r] * scale : -1e30f;
                float x1 = (km1 > 0.0f) ? s1[r] * scale : -1e30f;
                float rm = wred_max16(fmaxf(x0, x1));
                float mnew = fmaxf(mi[r], rm);
                float f = expf(mi[r] - mnew);
                float p0 = expf(x0 - mnew);
                float p1 = expf(x1 - mnew);
                float rs = wred_add16(p0 + p1);
                li[r] = li[r] * f + rs;
                mi[r] = mnew;
                acc0[r] *= f;
                acc1[r] *= f;
                Ps[w][r + 8 * g][ln] = f2bf(p0);
                Ps[w][r + 8 * g][16 + ln] = f2bf(p1);
            }
        }
        __syncthreads();

        if (active) {
            const bf16_t* pr = &Ps[w][ln][0];
            const bf16_t* v0p = &Vt[ln][g * 16];
            const bf16_t* v1p = &Vt[16 + ln][g * 16];
            v16bf pf = ldfrag(pr + g * 16, pr + g * 16 + 8);
            v16bf vf0 = ldfrag(v0p, v0p + 8);
            v16bf vf1 = ldfrag(v1p, v1p + 8);
            acc0 = __builtin_amdgcn_wmma_f32_16x16x32_bf16(
                false, pf, false, vf0, (short)0, acc0, false, false);
            acc1 = __builtin_amdgcn_wmma_f32_16x16x32_bf16(
                false, pf, false, vf1, (short)0, acc1, false, false);
        }
    }

    if (active) {
#pragma unroll
        for (int r = 0; r < 8; ++r) {
            int m = r + 8 * g;
            float inv = 1.0f / li[r];
            long long base = (tok0 + qt * 16 + m) * E_DIM + hh * 32;
            ctx[base + ln] = f2bf(acc0[r] * inv);
            ctx[base + 16 + ln] = f2bf(acc1[r] * inv);
        }
    }
}

// ---------------------------------------------------------------------------
// h = LayerNorm(x + y) * g + b ; writes f32 + bf16 (one wave per token)
// ---------------------------------------------------------------------------
__global__ __launch_bounds__(256)
void add_ln_k(const float* xin, const float* yin,
              const float* __restrict__ gw, const float* __restrict__ bw,
              float* hout, bf16_t* __restrict__ hbf, long long T) {
    int wave = threadIdx.x >> 5, lane = threadIdx.x & 31;
    long long tok = (long long)blockIdx.x * 8 + wave;
    if (tok >= T) return;
    const float* xp = xin + tok * E_DIM;
    const float* yp = yin + tok * E_DIM;
    float v[8];
    float s = 0.0f;
#pragma unroll
    for (int i = 0; i < 8; ++i) {
        v[i] = xp[lane + i * 32] + yp[lane + i * 32];
        s += v[i];
    }
    float mean = wred_add32(s) * (1.0f / E_DIM);
    float vs = 0.0f;
#pragma unroll
    for (int i = 0; i < 8; ++i) {
        float d = v[i] - mean;
        vs += d * d;
    }
    float var = wred_add32(vs) * (1.0f / E_DIM);
    float rstd = rsqrtf(var + 1e-5f);
#pragma unroll
    for (int i = 0; i < 8; ++i) {
        int e = lane + i * 32;
        float o = (v[i] - mean) * rstd * gw[e] + bw[e];
        hout[tok * E_DIM + e] = o;
        hbf[tok * E_DIM + e] = f2bf(o);
    }
}

// ---------------------------------------------------------------------------
// mixed = (mx*emb*mask + (1-mx)*h)*mask ; L2-normalize ; write bf16
// ---------------------------------------------------------------------------
__global__ __launch_bounds__(256)
void mixnorm_k(const float* __restrict__ emb, const float* __restrict__ mask,
               const float* __restrict__ h, const float* __restrict__ mixp,
               bf16_t* __restrict__ outbf, long long T) {
    int wave = threadIdx.x >> 5, lane = threadIdx.x & 31;
    long long tok = (long long)blockIdx.x * 8 + wave;
    if (tok >= T) return;
    float mx = mixp[0];
    float mk = mask[tok];
    float v[8];
    float ss = 0.0f;
#pragma unroll
    for (int i = 0; i < 8; ++i) {
        long long idx = tok * E_DIM + lane + i * 32;
        float x = emb[idx] * mk;
        float val = (mx * x + (1.0f - mx) * h[idx]) * mk;
        v[i] = val;
        ss += val * val;
    }
    float nrm = sqrtf(wred_add32(ss));
    float inv = 1.0f / (nrm + 1e-13f);
#pragma unroll
    for (int i = 0; i < 8; ++i)
        outbf[tok * E_DIM + lane + i * 32] = f2bf(v[i] * inv);
}

// ---------------------------------------------------------------------------
// d_stop = ms*emb + (1-ms)*h -> bf16
// ---------------------------------------------------------------------------
__global__ void dstop_k(const float* __restrict__ emb, const float* __restrict__ h,
                        const float* __restrict__ msp, bf16_t* __restrict__ out,
                        long long total) {
    long long i = (long long)blockIdx.x * 256 + threadIdx.x;
    if (i >= total) return;
    float ms = msp[0];
    out[i] = f2bf(ms * emb[i] + (1.0f - ms) * h[i]);
}

// ---------------------------------------------------------------------------
// sw = relu(t @ sw2^T + b) * dmask ; also to d_out (one wave per token)
// ---------------------------------------------------------------------------
__global__ __launch_bounds__(256)
void sw_k(const float* __restrict__ tbuf, const float* __restrict__ w2,
          const float* __restrict__ b2, const float* __restrict__ dmask,
          float* __restrict__ sw, float* __restrict__ outsw, long long T) {
    int wave = threadIdx.x >> 5, lane = threadIdx.x & 31;
    long long tok = (long long)blockIdx.x * 8 + wave;
    if (tok >= T) return;
    float s = 0.0f;
    for (int e = lane; e < 100; e += 32) s += tbuf[tok * 100 + e] * w2[e];
    s = wred_add32(s);
    if (lane == 0) {
        float v = fmaxf(s + b2[0], 0.0f) * dmask[tok];
        sw[tok] = v;
        outsw[tok] = v;
    }
}

// ---------------------------------------------------------------------------
// kernel pooling: pkqlog[b,q,k] = log(clip(sum_d exp(-(cos-mu)^2/2s^2)*sw * a))*qmask
// ---------------------------------------------------------------------------
__global__ __launch_bounds__(256)
void pool_k(const float* __restrict__ cosb, const float* __restrict__ sw,
            const float* __restrict__ mu, const float* __restrict__ sigma,
            const float* __restrict__ alpha, const float* __restrict__ qmask,
            float* __restrict__ pkqlog) {
    int bq = blockIdx.x;                 // b*32 + q
    int b = bq >> 5;
    float accv[NKER] = {};
    for (int d = threadIdx.x; d < D_LEN; d += 256) {
        float c = cosb[(long long)bq * D_LEN + d];
        float w = sw[(long long)b * D_LEN + d];
#pragma unroll
        for (int k = 0; k < NKER; ++k) {
            float diff = c - mu[k];
            float inv2 = 0.5f / (sigma[k] * sigma[k]);
            accv[k] += expf(-diff * diff * inv2) * w;
        }
    }
    __shared__ float sacc[256];
    for (int k = 0; k < NKER; ++k) {
        sacc[threadIdx.x] = accv[k];
        __syncthreads();
        for (int s = 128; s > 0; s >>= 1) {
            if (threadIdx.x < s) sacc[threadIdx.x] += sacc[threadIdx.x + s];
            __syncthreads();
        }
        if (threadIdx.x == 0) {
            float pkq = sacc[0] * alpha[k];
            pkqlog[(long long)bq * NKER + k] =
                logf(fmaxf(pkq, 1e-10f)) * qmask[bq];
        }
        __syncthreads();
    }
}

// ---------------------------------------------------------------------------
// score[b] = sum_{q,k} pkqlog[b,q,k] * kernel_w[k]
// ---------------------------------------------------------------------------
__global__ __launch_bounds__(32)
void score_k(const float* __restrict__ pkqlog, const float* __restrict__ kw,
             float* __restrict__ out) {
    int b = blockIdx.x;
    int lane = threadIdx.x;
    float s = 0.0f;
    for (int i = lane; i < Q_LEN * NKER; i += 32)
        s += pkqlog[(long long)b * Q_LEN * NKER + i] * kw[i % NKER];
    s = wred_add32(s);
    if (lane == 0) out[b] = s;
}

// ---------------------------------------------------------------------------
// host orchestration
// ---------------------------------------------------------------------------
extern "C" void kernel_launch(void* const* d_in, const int* in_sizes, int n_in,
                              void* d_out, int out_size, void* d_ws, size_t ws_size,
                              hipStream_t stream) {
    (void)in_sizes; (void)out_size; (void)ws_size;
    if (n_in < 26) return;

    const float* q_emb = (const float*)d_in[0];
    const float* d_emb = (const float*)d_in[1];
    const float* q_mask = (const float*)d_in[2];
    const float* d_mask = (const float*)d_in[3];
    const float* ipw = (const float*)d_in[4];
    const float* ipb = (const float*)d_in[5];
    const float* ow = (const float*)d_in[6];
    const float* ob = (const float*)d_in[7];
    const float* l1g = (const float*)d_in[8];
    const float* l1b = (const float*)d_in[9];
    const float* l2g = (const float*)d_in[10];
    const float* l2b = (const float*)d_in[11];
    const float* f1w = (const float*)d_in[12];
    const float* f1b = (const float*)d_in[13];
    const float* f2w = (const float*)d_in[14];
    const float* f2b = (const float*)d_in[15];
    const float* mixer = (const float*)d_in[16];
    const float* mixer_stop = (const float*)d_in[17];
    const float* mu = (const float*)d_in[18];
    const float* sigma = (const float*)d_in[19];
    const float* kw = (const float*)d_in[20];
    const float* alpha = (const float*)d_in[21];
    const float* sw1w = (const float*)d_in[22];
    const float* sw1b = (const float*)d_in[23];
    const float* sw2w = (const float*)d_in[24];
    const float* sw2b = (const float*)d_in[25];
    float* out = (float*)d_out;

    // ---- workspace carving (256B aligned regions) ----
    size_t off = 0;
    char* base = (char*)d_ws;
    auto carve = [&](size_t bytes) -> void* {
        void* p = base + off;
        off += (bytes + 255) & ~(size_t)255;
        return p;
    };
    float*  pos    = (float*)carve((size_t)512 * E_DIM * 4);
    bf16_t* ipw_bf = (bf16_t*)carve((size_t)2 * 768 * 256 * 2);
    bf16_t* ow_bf  = (bf16_t*)carve((size_t)2 * 256 * 256 * 2);
    bf16_t* f1w_bf = (bf16_t*)carve((size_t)2 * 1024 * 256 * 2);
    bf16_t* f2w_bf = (bf16_t*)carve((size_t)2 * 256 * 1024 * 2);
    bf16_t* sw1_bf = (bf16_t*)carve((size_t)100 * 256 * 2);
    float*  h      = (float*)carve((size_t)TD * E_DIM * 4);
    bf16_t* hbf    = (bf16_t*)carve((size_t)TD * E_DIM * 2);
    bf16_t* qkvbf  = (bf16_t*)carve((size_t)TD * 768 * 2);
    bf16_t* ctxbf  = (bf16_t*)carve((size_t)TD * E_DIM * 2);
    float*  gout   = (float*)carve((size_t)TD * E_DIM * 4);
    bf16_t* ff1bf  = (bf16_t*)carve((size_t)TD * FF_DIM * 2);
    bf16_t* qnbf   = (bf16_t*)carve((size_t)TQ * E_DIM * 2);
    float*  swb    = (float*)carve((size_t)TD * 4);
    float*  pkql   = (float*)carve((size_t)B_SZ * Q_LEN * NKER * 4);
    // reuse (sequencing makes these safe):
    bf16_t* dnbf    = hbf;               // after doc layers, hbf is free
    bf16_t* dstopbf = ctxbf;             // after doc layers, ctx is free
    float*  cosb    = gout;              // 16.8 MB at region base
    float*  tbuf    = gout + 8388608;    // +32 MB (floats), 52.4 MB used

    auto conv = [&](const float* s, bf16_t* dst, long long n) {
        f2bf_k<<<(unsigned)((n + 255) / 256), 256, 0, stream>>>(s, dst, n);
    };
    conv(ipw, ipw_bf, 2LL * 768 * 256);
    conv(ow, ow_bf, 2LL * 256 * 256);
    conv(f1w, f1w_bf, 2LL * 1024 * 256);
    conv(f2w, f2w_bf, 2LL * 256 * 1024);
    conv(sw1w, sw1_bf, 100LL * 256);
    posenc_k<<<(512 * 128 + 255) / 256, 256, 0, stream>>>(pos);

    auto run_path = [&](const float* emb, const float* mask, int S, long long T) {
        long long total = T * E_DIM;
        prep_k<<<(unsigned)((total + 255) / 256), 256, 0, stream>>>(
            emb, mask, pos, h, hbf, S, total);
        unsigned mt = (unsigned)((T + 127) / 128);
        unsigned qtiles = (unsigned)((S / 16 + 3) / 4);   // 4 q-tiles per block
        for (int l = 0; l < 2; ++l) {
            // QKV projection -> bf16
            gemm_bt<<<dim3(768 / 64, mt, 1), 256, 0, stream>>>(
                hbf, 0, E_DIM, ipw_bf + (long long)l * 768 * 256, 0, E_DIM,
                ipb + (long long)l * 768, (float*)0, qkvbf, 0, 768,
                (int)T, 768, E_DIM, 0);
            // flash attention (K/V tile shared by 4 waves)
            attn_flash<<<dim3(qtiles, NHEAD, B_SZ), 128, 0, stream>>>(
                qkvbf, mask, ctxbf, S);
            // output projection -> f32
            gemm_bt<<<dim3(E_DIM / 64, mt, 1), 256, 0, stream>>>(
                ctxbf, 0, E_DIM, ow_bf + (long long)l * 256 * 256, 0, E_DIM,
                ob + (long long)l * 256, gout, (bf16_t*)0, 0, E_DIM,
                (int)T, E_DIM, E_DIM, 0);
            add_ln_k<<<(unsigned)((T + 7) / 8), 256, 0, stream>>>(
                h, gout, l1g + (long long)l * 256, l1b + (long long)l * 256,
                h, hbf, T);
            // FF1 (relu) -> bf16
            gemm_bt<<<dim3(FF_DIM / 64, mt, 1), 256, 0, stream>>>(
                hbf, 0, E_DIM, f1w_bf + (long long)l * 1024 * 256, 0, E_DIM,
                f1b + (long long)l * 1024, (float*)0, ff1bf, 0, FF_DIM,
                (int)T, FF_DIM, E_DIM, 1);
            // FF2 -> f32
            gemm_bt<<<dim3(E_DIM / 64, mt, 1), 256, 0, stream>>>(
                ff1bf, 0, FF_DIM, f2w_bf + (long long)l * 256 * 1024, 0, FF_DIM,
                f2b + (long long)l * 256, gout, (bf16_t*)0, 0, E_DIM,
                (int)T, E_DIM, FF_DIM, 0);
            add_ln_k<<<(unsigned)((T + 7) / 8), 256, 0, stream>>>(
                h, gout, l2g + (long long)l * 256, l2b + (long long)l * 256,
                h, hbf, T);
        }
    };

    // ---- query path ----
    run_path(q_emb, q_mask, Q_LEN, TQ);
    mixnorm_k<<<TQ / 8, 256, 0, stream>>>(q_emb, q_mask, h, mixer, qnbf, TQ);

    // ---- document path ----
    run_path(d_emb, d_mask, D_LEN, TD);
    mixnorm_k<<<TD / 8, 256, 0, stream>>>(d_emb, d_mask, h, mixer, dnbf, TD);
    dstop_k<<<(unsigned)(((long long)TD * E_DIM + 255) / 256), 256, 0, stream>>>(
        d_emb, h, mixer_stop, dstopbf, (long long)TD * E_DIM);

    // ---- stopword gate: tanh MLP then relu-dot ----
    gemm_bt<<<dim3(2, (TD + 127) / 128, 1), 256, 0, stream>>>(
        dstopbf, 0, E_DIM, sw1_bf, 0, E_DIM, sw1b, tbuf, (bf16_t*)0, 0, 100,
        TD, 100, E_DIM, 2);
    sw_k<<<TD / 8, 256, 0, stream>>>(tbuf, sw2w, sw2b, d_mask, swb, out + B_SZ, TD);

    // ---- cosine similarity (batched GEMM): cos[b,32,512] ----
    gemm_bt<<<dim3(D_LEN / 64, 1, B_SZ), 256, 0, stream>>>(
        qnbf, (long long)Q_LEN * E_DIM, E_DIM,
        dnbf, (long long)D_LEN * E_DIM, E_DIM,
        (float*)0, cosb, (bf16_t*)0, (long long)Q_LEN * D_LEN, D_LEN,
        Q_LEN, D_LEN, E_DIM, 0);

    // ---- kernel pooling + score ----
    pool_k<<<B_SZ * Q_LEN, 256, 0, stream>>>(cosb, swb, mu, sigma, alpha,
                                             q_mask, pkql);
    score_k<<<B_SZ, 32, 0, stream>>>(pkql, kw, out);
}
